// LatentVoxelGrid_85186381348960
// MI455X (gfx1250) — matrix operand order repaired and checked by reading, not compiled, not run on hardware
//
#include <hip/hip_runtime.h>
#include <hip/hip_bf16.h>

#define N_PTS 65536
#define M_VOX 131072
#define KCAND 8
#define DDIM  64
#define TAU_F 0.3f
#define EPS_F 1e-6f

typedef __attribute__((ext_vector_type(16))) __bf16 v16bf;
typedef __attribute__((ext_vector_type(8)))  float  v8f;

__device__ __forceinline__ float sigmoidf_(float x) { return 1.0f / (1.0f + __expf(-x)); }

// Pack A-matrix lane data: elems 0..7 = src[0..7], 8..15 = src[16..23]
__device__ __forceinline__ v16bf pack_a16(const float* __restrict__ src) {
    const float4 q0 = *(const float4*)(src + 0);
    const float4 q1 = *(const float4*)(src + 4);
    const float4 q2 = *(const float4*)(src + 16);
    const float4 q3 = *(const float4*)(src + 20);
    v16bf a;
    a[0] = (__bf16)q0.x; a[1] = (__bf16)q0.y; a[2]  = (__bf16)q0.z; a[3]  = (__bf16)q0.w;
    a[4] = (__bf16)q1.x; a[5] = (__bf16)q1.y; a[6]  = (__bf16)q1.z; a[7]  = (__bf16)q1.w;
    a[8] = (__bf16)q2.x; a[9] = (__bf16)q2.y; a[10] = (__bf16)q2.z; a[11] = (__bf16)q2.w;
    a[12] = (__bf16)q3.x; a[13] = (__bf16)q3.y; a[14] = (__bf16)q3.z; a[15] = (__bf16)q3.w;
    return a;
}

__device__ __forceinline__ v16bf pack_a16_scaled(const float* __restrict__ src, float s) {
    const float4 q0 = *(const float4*)(src + 0);
    const float4 q1 = *(const float4*)(src + 4);
    const float4 q2 = *(const float4*)(src + 16);
    const float4 q3 = *(const float4*)(src + 20);
    v16bf a;
    a[0] = (__bf16)(q0.x*s); a[1] = (__bf16)(q0.y*s); a[2]  = (__bf16)(q0.z*s); a[3]  = (__bf16)(q0.w*s);
    a[4] = (__bf16)(q1.x*s); a[5] = (__bf16)(q1.y*s); a[6]  = (__bf16)(q1.z*s); a[7]  = (__bf16)(q1.w*s);
    a[8] = (__bf16)(q2.x*s); a[9] = (__bf16)(q2.y*s); a[10] = (__bf16)(q2.z*s); a[11] = (__bf16)(q2.w*s);
    a[12] = (__bf16)(q3.x*s); a[13] = (__bf16)(q3.y*s); a[14] = (__bf16)(q3.z*s); a[15] = (__bf16)(q3.w*s);
    return a;
}

// Pack 16 contiguous floats (for K-contiguous weight rows: W[col*64 + K])
__device__ __forceinline__ v16bf pack16_contig(const float* __restrict__ src) {
    const float4 q0 = *(const float4*)(src + 0);
    const float4 q1 = *(const float4*)(src + 4);
    const float4 q2 = *(const float4*)(src + 8);
    const float4 q3 = *(const float4*)(src + 12);
    v16bf a;
    a[0] = (__bf16)q0.x; a[1] = (__bf16)q0.y; a[2]  = (__bf16)q0.z; a[3]  = (__bf16)q0.w;
    a[4] = (__bf16)q1.x; a[5] = (__bf16)q1.y; a[6]  = (__bf16)q1.z; a[7]  = (__bf16)q1.w;
    a[8] = (__bf16)q2.x; a[9] = (__bf16)q2.y; a[10] = (__bf16)q2.z; a[11] = (__bf16)q2.w;
    a[12] = (__bf16)q3.x; a[13] = (__bf16)q3.y; a[14] = (__bf16)q3.z; a[15] = (__bf16)q3.w;
    return a;
}

// ---------------------------------------------------------------------------
// Kernel 0: zero the scatter scratch (msg[M*64] ++ wsum[M]) each call.
// ---------------------------------------------------------------------------
__global__ void lvg_zero_kernel(float4* __restrict__ p, int count4) {
    int i = blockIdx.x * blockDim.x + threadIdx.x;
    int stride = gridDim.x * blockDim.x;
    const float4 z = make_float4(0.f, 0.f, 0.f, 0.f);
    for (; i < count4; i += stride) p[i] = z;
}

// ---------------------------------------------------------------------------
// Kernel 1: similarity MLP (WMMA bf16) + softmax over K + atomic scatter.
// One wave-tile = 16 rows = 2 points x 8 candidates. B frags staged in LDS
// (compiler hoists them to registers across the tile loop).
// ---------------------------------------------------------------------------
#define K1_FRAGS 20
__global__ __launch_bounds__(128)
void lvg_sim_scatter_kernel(const float* __restrict__ f_pts,
                            const float* __restrict__ pts,
                            const float* __restrict__ z_latent,
                            const float* __restrict__ centers,
                            const int*   __restrict__ cand_idx,
                            const float* __restrict__ sim_w1,
                            const float* __restrict__ sim_b1,
                            const float* __restrict__ sim_w2,
                            const float* __restrict__ sim_b2,
                            float* __restrict__ msg,
                            float* __restrict__ wsum)
{
    __shared__ v16bf sB[K1_FRAGS * 32];          // 20 KB

    const int lane = threadIdx.x & 31;
    const int wave = threadIdx.x >> 5;
    const int half = lane >> 4;
    const int n    = lane & 15;
    const int r    = lane & 15;

    // ---- stage packed B fragments of sim_w1 (131x64, K padded to 160) once per block
    for (int f = wave; f < K1_FRAGS; f += 4) {
        const int kc = f >> 2, t = f & 3;
        v16bf b;
#pragma unroll
        for (int e = 0; e < 16; ++e) {
            const int K = kc * 32 + (e < 8 ? (half ? 8 : 0) + e : 16 + (half ? 8 : 0) + (e - 8));
            b[e] = (__bf16)((K < 131) ? sim_w1[K * DDIM + t * 16 + n] : 0.0f);
        }
        sB[f * 32 + lane] = b;
    }
    __syncthreads();

    // ---- tile-invariant hoists
    float b1h[4], w2h[4];
#pragma unroll
    for (int t = 0; t < 4; ++t) { b1h[t] = sim_b1[t * 16 + n]; w2h[t] = sim_w2[t * 16 + n]; }
    const float b2 = sim_b2[0];
    const float inv_tau = 1.0f / TAU_F;

    const int wv       = blockIdx.x * 4 + wave;
    const int wstride  = gridDim.x * 4;
    const int n_tiles  = (N_PTS * KCAND) / 16;   // 32768

#pragma unroll 1
    for (int tile = wv; tile < n_tiles; tile += wstride) {
        const int base_pair = tile * 16;
        const int pr = (base_pair + r) >> 3;     // point of row r
        const int kr = (base_pair + r) & 7;
        const int cr = cand_idx[pr * KCAND + kr];

        // --- A fragments: x = [f_pts(64) | z_c(64) | delta(3) | pad -> 160]
        v16bf a[5];
        {
            const float* fsrc = f_pts + pr * DDIM;
            const float* zsrc = z_latent + cr * DDIM;
            const int off = half ? 8 : 0;
            a[0] = pack_a16(fsrc + off);
            a[1] = pack_a16(fsrc + 32 + off);
            a[2] = pack_a16(zsrc + off);
            a[3] = pack_a16(zsrc + 32 + off);
            v16bf a4;
#pragma unroll
            for (int e = 0; e < 16; ++e) a4[e] = (__bf16)0.0f;
            if (half == 0) {
#pragma unroll
                for (int j = 0; j < 3; ++j)
                    a4[j] = (__bf16)(pts[pr * 3 + j] - centers[cr * 3 + j]);
            }
            a[4] = a4;
        }

        // --- layer 1: 5 x 4 WMMAs, B from LDS (loop-invariant -> reg-resident)
        v8f c[4];
#pragma unroll
        for (int t = 0; t < 4; ++t) {
#pragma unroll
            for (int v = 0; v < 8; ++v) c[t][v] = b1h[t];
        }
#pragma unroll
        for (int kc = 0; kc < 5; ++kc) {
#pragma unroll
            for (int t = 0; t < 4; ++t) {
                const v16bf b = sB[(kc * 4 + t) * 32 + lane];
                c[t] = __builtin_amdgcn_wmma_f32_16x16x32_bf16(
                    false, a[kc], false, b, (short)0, c[t], false, false);
            }
        }

        // --- ReLU + layer 2 (64->1): lane-local dot + 16-lane xor-tree
        float part[8];
#pragma unroll
        for (int v = 0; v < 8; ++v) part[v] = 0.0f;
#pragma unroll
        for (int t = 0; t < 4; ++t) {
#pragma unroll
            for (int v = 0; v < 8; ++v) part[v] += fmaxf(c[t][v], 0.0f) * w2h[t];
        }
#pragma unroll
        for (int off = 1; off < 16; off <<= 1) {
#pragma unroll
            for (int v = 0; v < 8; ++v) part[v] += __shfl_xor(part[v], off, 32);
        }

        // --- softmax over K=8 (each half-wave holds one point's logits)
        float mx = -1e30f;
#pragma unroll
        for (int v = 0; v < 8; ++v) { part[v] = (part[v] + b2) * inv_tau; mx = fmaxf(mx, part[v]); }
        float s = 0.0f;
        float w[8];
#pragma unroll
        for (int v = 0; v < 8; ++v) { w[v] = __expf(part[v] - mx); s += w[v]; }
        const float invs = 1.0f / s;

        // --- scatter: half h handles point tile*2+h; lane adds 4 contiguous dims
        const int p = tile * 2 + half;
        const float4 f4 = *(const float4*)(f_pts + p * DDIM + n * 4);
#pragma unroll
        for (int v = 0; v < 8; ++v) {
            const float wv_ = w[v] * invs;
            const int cv = cand_idx[p * KCAND + v];
            float* mrow = msg + cv * DDIM + n * 4;
            atomicAdd(mrow + 0, wv_ * f4.x);
            atomicAdd(mrow + 1, wv_ * f4.y);
            atomicAdd(mrow + 2, wv_ * f4.z);
            atomicAdd(mrow + 3, wv_ * f4.w);
            if (n == 0) atomicAdd(wsum + cv, wv_);
        }
    }
}

// ---------------------------------------------------------------------------
// Kernel 2: per-voxel gate MLP + GRUCell via WMMA bf16, B frags staged in LDS.
// Frags: [0..15] gate_w1 (kc*4+t), [16..39] W_ih, [40..63] W_hh ((t*3+blk)*2+kc)
// ---------------------------------------------------------------------------
#define K2_FRAGS 64
__global__ __launch_bounds__(128)
void lvg_voxel_update_kernel(const float* __restrict__ z_latent,
                             const float* __restrict__ msg,
                             const float* __restrict__ wsum,
                             const float* __restrict__ gate_w1,
                             const float* __restrict__ gate_b1,
                             const float* __restrict__ gate_w2,
                             const float* __restrict__ gate_b2,
                             const float* __restrict__ W_ih,
                             const float* __restrict__ W_hh,
                             const float* __restrict__ b_ih,
                             const float* __restrict__ b_hh,
                             float* __restrict__ out)
{
    __shared__ v16bf sB[K2_FRAGS * 32];          // 64 KB

    const int lane = threadIdx.x & 31;
    const int wave = threadIdx.x >> 5;
    const int half = lane >> 4;
    const int n    = lane & 15;
    const int r    = lane & 15;

    // ---- stage packed B fragments once per block
    for (int f = wave; f < K2_FRAGS; f += 4) {
        v16bf b;
        if (f < 16) {                            // gate_w1 (128 x 64), K-strided
            const int kc = f >> 2, t = f & 3;
#pragma unroll
            for (int e = 0; e < 16; ++e) {
                const int K = kc * 32 + (e < 8 ? (half ? 8 : 0) + e : 16 + (half ? 8 : 0) + (e - 8));
                b[e] = (__bf16)gate_w1[K * DDIM + t * 16 + n];
            }
        } else {                                 // W_ih / W_hh: B[k][j]=W[j*64+k], K-contiguous
            const int g   = (f < 40) ? (f - 16) : (f - 40);
            const float* W = (f < 40) ? W_ih : W_hh;
            const int kc  = g & 1;
            const int tb  = g >> 1;
            const int t   = tb / 3;
            const int blk = tb % 3;
            const int col = blk * DDIM + t * 16 + n;
            b = pack_a16(W + col * DDIM + kc * 32 + (half ? 8 : 0));
        }
        sB[f * 32 + lane] = b;
    }
    __syncthreads();

    // ---- tile-invariant hoists
    float gb1h[4], gw2h[4];
#pragma unroll
    for (int t = 0; t < 4; ++t) { gb1h[t] = gate_b1[t * 16 + n]; gw2h[t] = gate_w2[t * 16 + n]; }
    const float gb2 = gate_b2[0];
    float bih_h[12], bhh_h[12];
#pragma unroll
    for (int t = 0; t < 4; ++t) {
#pragma unroll
        for (int blk = 0; blk < 3; ++blk) {
            const int col = blk * DDIM + t * 16 + n;
            bih_h[t * 3 + blk] = b_ih[col];
            bhh_h[t * 3 + blk] = b_hh[col];
        }
    }

    const int wv      = blockIdx.x * 4 + wave;
    const int wstride = gridDim.x * 4;
    const int n_tiles = M_VOX / 16;              // 8192

#pragma unroll 1
    for (int tile = wv; tile < n_tiles; tile += wstride) {
        const int base  = tile * 16;
        const int row_r = base + r;
        const float inv_w = 1.0f / (wsum[row_r] + EPS_F);

        // --- A fragments: normalized msg and z_latent
        v16bf a_m[2], a_z[2];
        {
            const float* msrc = msg + row_r * DDIM;
            const float* zsrc = z_latent + row_r * DDIM;
            const int off = half ? 8 : 0;
            a_m[0] = pack_a16_scaled(msrc + off,      inv_w);
            a_m[1] = pack_a16_scaled(msrc + 32 + off, inv_w);
            a_z[0] = pack_a16(zsrc + off);
            a_z[1] = pack_a16(zsrc + 32 + off);
        }

        // --- gate: sigmoid(relu([z|msg] @ gate_w1 + b1) @ gate_w2 + b2)
        float gp[8];
#pragma unroll
        for (int v = 0; v < 8; ++v) gp[v] = 0.0f;
#pragma unroll
        for (int t = 0; t < 4; ++t) {
            v8f c;
#pragma unroll
            for (int v = 0; v < 8; ++v) c[v] = gb1h[t];
#pragma unroll
            for (int kc = 0; kc < 4; ++kc) {
                const v16bf b  = sB[(kc * 4 + t) * 32 + lane];
                const v16bf af = (kc < 2) ? a_z[kc & 1] : a_m[kc & 1];
                c = __builtin_amdgcn_wmma_f32_16x16x32_bf16(
                    false, af, false, b, (short)0, c, false, false);
            }
#pragma unroll
            for (int v = 0; v < 8; ++v) gp[v] += fmaxf(c[v], 0.0f) * gw2h[t];
        }
#pragma unroll
        for (int off = 1; off < 16; off <<= 1) {
#pragma unroll
            for (int v = 0; v < 8; ++v) gp[v] += __shfl_xor(gp[v], off, 32);
        }
        float gate_v[8];
#pragma unroll
        for (int v = 0; v < 8; ++v) gate_v[v] = sigmoidf_(gp[v] + gb2);

        // --- GRUCell, 16 output columns per step (r/z/n tiles layout-aligned)
#pragma unroll
        for (int t = 0; t < 4; ++t) {
            v8f ci[3], ch[3];
#pragma unroll
            for (int blk = 0; blk < 3; ++blk) {
#pragma unroll
                for (int v = 0; v < 8; ++v) {
                    ci[blk][v] = bih_h[t * 3 + blk];
                    ch[blk][v] = bhh_h[t * 3 + blk];
                }
#pragma unroll
                for (int kc = 0; kc < 2; ++kc) {
                    const int fi = 16 + (t * 3 + blk) * 2 + kc;
                    const v16bf bI = sB[fi * 32 + lane];
                    const v16bf bH = sB[(fi + 24) * 32 + lane];
                    ci[blk] = __builtin_amdgcn_wmma_f32_16x16x32_bf16(
                        false, a_m[kc], false, bI, (short)0, ci[blk], false, false);
                    ch[blk] = __builtin_amdgcn_wmma_f32_16x16x32_bf16(
                        false, a_z[kc], false, bH, (short)0, ch[blk], false, false);
                }
            }
#pragma unroll
            for (int v = 0; v < 8; ++v) {
                const int row  = base + v + (half ? 8 : 0);
                const int cidx = row * DDIM + t * 16 + n;
                const float zv = z_latent[cidx];
                const float rr = sigmoidf_(ci[0][v] + ch[0][v]);
                const float zz = sigmoidf_(ci[1][v] + ch[1][v]);
                const float nn = tanhf(ci[2][v] + rr * ch[2][v]);
                const float hn = (1.0f - zz) * nn + zz * zv;
                out[cidx] = zv + gate_v[v] * (hn - zv);
            }
        }
    }
}

// ---------------------------------------------------------------------------
extern "C" void kernel_launch(void* const* d_in, const int* in_sizes, int n_in,
                              void* d_out, int out_size, void* d_ws, size_t ws_size,
                              hipStream_t stream)
{
    const float* f_pts    = (const float*)d_in[0];
    const float* pts      = (const float*)d_in[1];
    const float* z_latent = (const float*)d_in[2];
    const float* centers  = (const float*)d_in[3];
    const int*   cand_idx = (const int*)  d_in[4];
    const float* sim_w1   = (const float*)d_in[5];
    const float* sim_b1   = (const float*)d_in[6];
    const float* sim_w2   = (const float*)d_in[7];
    const float* sim_b2   = (const float*)d_in[8];
    const float* gate_w1  = (const float*)d_in[9];
    const float* gate_b1  = (const float*)d_in[10];
    const float* gate_w2  = (const float*)d_in[11];
    const float* gate_b2  = (const float*)d_in[12];
    const float* W_ih     = (const float*)d_in[13];
    const float* W_hh     = (const float*)d_in[14];
    const float* b_ih     = (const float*)d_in[15];
    const float* b_hh     = (const float*)d_in[16];

    float* msg  = (float*)d_ws;                 // M*64 floats
    float* wsum = msg + (size_t)M_VOX * DDIM;   // M floats
    const int zero_count4 = (M_VOX * DDIM + M_VOX) / 4;

    hipLaunchKernelGGL(lvg_zero_kernel, dim3(2048), dim3(256), 0, stream,
                       (float4*)msg, zero_count4);

    hipLaunchKernelGGL(lvg_sim_scatter_kernel, dim3(2048), dim3(128), 0, stream,
                       f_pts, pts, z_latent, centers, cand_idx,
                       sim_w1, sim_b1, sim_w2, sim_b2, msg, wsum);

    hipLaunchKernelGGL(lvg_voxel_update_kernel, dim3(512), dim3(128), 0, stream,
                       z_latent, msg, wsum,
                       gate_w1, gate_b1, gate_w2, gate_b2,
                       W_ih, W_hh, b_ih, b_hh, (float*)d_out);
}
